// GlobalPtrHead_80461917323645
// MI455X (gfx1250) — matrix operand order repaired and checked by reading, not compiled, not run on hardware
//
#include <hip/hip_runtime.h>

// ---------------- types ----------------
typedef __attribute__((ext_vector_type(16))) __bf16 v16bf;
typedef __attribute__((ext_vector_type(8)))  __bf16 v8bf;
typedef __attribute__((ext_vector_type(4)))  __bf16 v4bf;
typedef __attribute__((ext_vector_type(8)))  float  v8f;

typedef __attribute__((ext_vector_type(4))) unsigned tdm_uint4;
typedef __attribute__((ext_vector_type(8))) int      tdm_int8;
typedef __attribute__((ext_vector_type(4))) int      tdm_int4;

#define NUM_LABELS 9
#define PROJ 64
#define B_   8
#define L_   1024
#define HS_  1024
#define NTOT (2 * NUM_LABELS * PROJ)   // 1152

#if defined(__HIP_DEVICE_COMPILE__) && __has_builtin(__builtin_amdgcn_tensor_load_to_lds)
#define HAVE_TDM 1
#else
#define HAVE_TDM 0
#endif

__device__ __forceinline__ __bf16 f2bf(float f) { return (__bf16)f; }

// cheap, accurate-enough sin/cos for |x| <= ~1100 rad:
// round-to-nearest multiple of 2*pi (two-term Cody-Waite), then HW trans ops.
__device__ __forceinline__ void fast_sincos(float x, float* s, float* c) {
    float k = rintf(x * 0.15915494309189535f);       // x / (2*pi)
    float r = fmaf(k, -6.28318548f, x);              // hi part of 2*pi
    r = fmaf(k, 1.7484555e-07f, r);                  // lo part (2*pi - hi)
    *s = __sinf(r);
    *c = __cosf(r);
}

// =====================================================================
// Kernel 1: proj = HS @ W + b  (bf16 WMMA, f32 accumulate), then RoPE,
// store q/k as bf16 into workspace, layout [b][h][pos][d].
// One wave owns one 16(M) x 16(N) tile; the 8 waves of a block share a
// W tile that is converted to bf16 and stored into LDS **pre-swizzled
// into the per-lane WMMA B-fragment layout** (slot stride 20 halves ->
// dword index lane*10 mod 64: conflict-free across all 32 lanes).
// grid = (72, 64), block = 256
// =====================================================================
__global__ __launch_bounds__(256) void proj_rope_kernel(
    const float* __restrict__ hs,     // [8192][1024]
    const float* __restrict__ W,      // [1024][1152]
    const float* __restrict__ bias,   // [1152]
    unsigned short* __restrict__ qraw,
    unsigned short* __restrict__ kraw)
{
    __shared__ alignas(16) __bf16 ldsB[32 * 20];   // one 16-elem slot per lane (stride 20)

    const int lane = threadIdx.x & 31;
    const int wave = threadIdx.x >> 5;
    const int half = lane >> 4;
    const int nl   = lane & 15;

    const int n0 = blockIdx.x * 16;
    const int m0 = blockIdx.y * 128 + wave * 16;

    const int rowA = m0 + nl;                 // A row (M = lane&15 in both halves)
    const float* hrow = hs + (size_t)rowA * HS_;
    const int kbA = 8 * half;                 // A K-subgroup base

    const int r0 = threadIdx.x >> 4;          // cooperative-fill coords
    const int c0 = threadIdx.x & 15;

    v8f c = {};

    for (int kk = 0; kk < HS_; kk += 32) {
        __syncthreads();
        {   // W[kk+k][n0+c0] -> bf16, swizzled to B-fragment slots:
            // lane c0 + 16*h: j = k - 8h for k in 8h..8h+7, j = k - 8 - 8h for k in 16+8h..23+8h
            float w0 = W[(size_t)(kk + r0)      * NTOT + n0 + c0];
            float w1 = W[(size_t)(kk + r0 + 16) * NTOT + n0 + c0];
            ldsB[(c0 + 16 * (r0 >> 3)) * 20 + (r0 & 7)]     = f2bf(w0);
            ldsB[(c0 + 16 * (r0 >> 3)) * 20 + 8 + (r0 & 7)] = f2bf(w1);
        }
        __syncthreads();

        // A: 16x32 f32 -> bf16 per ISA lane layout
        float4 f0 = *(const float4*)(hrow + kk + kbA);
        float4 f1 = *(const float4*)(hrow + kk + kbA + 4);
        float4 f2 = *(const float4*)(hrow + kk + 16 + kbA);
        float4 f3 = *(const float4*)(hrow + kk + 16 + kbA + 4);
        v16bf a;
        a[0]  = f2bf(f0.x); a[1]  = f2bf(f0.y); a[2]  = f2bf(f0.z); a[3]  = f2bf(f0.w);
        a[4]  = f2bf(f1.x); a[5]  = f2bf(f1.y); a[6]  = f2bf(f1.z); a[7]  = f2bf(f1.w);
        a[8]  = f2bf(f2.x); a[9]  = f2bf(f2.y); a[10] = f2bf(f2.z); a[11] = f2bf(f2.w);
        a[12] = f2bf(f3.x); a[13] = f2bf(f3.y); a[14] = f2bf(f3.z); a[15] = f2bf(f3.w);

        // B fragment: 4x ds_load_b64 from this lane's slot
        const __bf16* slot = &ldsB[lane * 20];
        v4bf b0 = *(const v4bf*)(slot + 0);
        v4bf b1 = *(const v4bf*)(slot + 4);
        v4bf b2 = *(const v4bf*)(slot + 8);
        v4bf b3 = *(const v4bf*)(slot + 12);
        v16bf bm;
        #pragma unroll
        for (int i = 0; i < 4; ++i) {
            bm[i] = b0[i]; bm[4 + i] = b1[i]; bm[8 + i] = b2[i]; bm[12 + i] = b3[i];
        }

        c = __builtin_amdgcn_wmma_f32_16x16x32_bf16(
                false, a, false, bm, (short)0, c, false, false);
    }

    // ---------------- epilogue: bias + RoPE + bf16 store ----------------
    const int ncol   = n0 + nl;
    const int head   = ncol >> 7;
    const int within = ncol & 127;
    const int dd     = within & 63;
    __bf16* buf = (within & 64) ? (__bf16*)kraw : (__bf16*)qraw;

    const int bb = blockIdx.y >> 3;
    const float bv = bias[ncol];
    // 10000^(-(dd&~1)/64) = exp2(-log2(1e4)/64 * (dd&~1)); native v_exp_f32
    const float freq = exp2f(-0.20762050593046f * (float)(dd & ~1));
    const size_t base = ((size_t)(bb * NUM_LABELS + head) * L_) * PROJ + dd;

    #pragma unroll
    for (int r = 0; r < 8; ++r) {
        float v = c[r] + bv;
        float p = __shfl_xor(v, 1, 32);       // partner of the RoPE pair
        float x2 = (dd & 1) ? p : -p;
        int m   = m0 + r + 8 * half;
        int pos = m & (L_ - 1);
        float so, co;
        fast_sincos((float)pos * freq, &so, &co);
        buf[base + (size_t)pos * PROJ] = f2bf(v * co + x2 * so);
    }
}

// =====================================================================
// Kernel 2: logits[b,h,m,n] = (q[m]·k[n]) / 8, causal + key-pad mask.
// The block's shared K tile (16 n-rows x 64 d, bf16) is DMA'd into LDS
// by the Tensor Data Mover (wave 0 issues tensor_load_to_lds, waits on
// TENSORcnt, barrier publishes). D# uses pad_enable (4 dwords per 32)
// so LDS rows land at stride 72 elems (16B aligned, bank-friendly).
// One wave per 16x16 output tile, 2 bf16 WMMAs over d=64.
// grid = (64 n-tiles, 8 m-groups, 72 b*h), block = 256 (8 m-tiles)
// =====================================================================
__global__ __launch_bounds__(256) void qk_logits_kernel(
    const unsigned short* __restrict__ qraw,
    const unsigned short* __restrict__ kraw,
    const int* __restrict__ amask,    // [8][1024]
    float* __restrict__ out)          // [8][9][1024][1024]
{
    __shared__ alignas(16) __bf16 ldsK[16 * 72];   // padded rows: 64 data + 8 pad

    const int lane = threadIdx.x & 31;
    const int wave = threadIdx.x >> 5;
    const int half = lane >> 4;
    const int nl   = lane & 15;

    const int n0    = blockIdx.x * 16;
    const int mBase = blockIdx.y * 128;
    const int m0    = mBase + wave * 16;
    const int bh    = blockIdx.z;
    const int bb    = bh / NUM_LABELS;
    const int n     = n0 + nl;

    const size_t obase = (size_t)bh * L_ * L_;
    const float NEGINF = -__builtin_inff();

    const __bf16* qb = (const __bf16*)qraw + (size_t)bh * L_ * PROJ;
    const __bf16* kb = (const __bf16*)kraw + (size_t)bh * L_ * PROJ;

    const bool blockDead = (mBase > n0 + 15);   // uniform across the block
    if (!blockDead) {
#if HAVE_TDM
        if (wave == 0) {
            // ---- Tensor DMA descriptor: 2-D tensor [1024 rows][64 d] bf16,
            //      tile = rows n0..n0+15, all 64 d; LDS pad 4 dw per 32 dw.
            size_t   ga     = (size_t)(const void*)(kb + (size_t)n0 * PROJ);
            unsigned ldsoff = (unsigned)(size_t)(&ldsK[0]);   // low 32 bits = LDS offset
            tdm_uint4 g0 = { 1u,                               // count=1, user mode
                             ldsoff,
                             (unsigned)ga,
                             (unsigned)(ga >> 32) | (2u << 30) };  // type=2 "image"
            tdm_int8 g1 = {
                (int)((1u << 16)            // data_size = 2 bytes
                    | (1u << 20)            // pad_enable
                    | (4u << 22)            // pad_interval: every 32 dwords
                    | (3u << 25)),          // pad_amount: 4 dwords
                (int)(64u   << 16),         // tensor_dim0 = 64 (lo16)
                (int)(1024u << 16),         // dim0 hi16 = 0 | tensor_dim1 = 1024 (lo16)
                (int)(64u   << 16),         // dim1 hi16 = 0 | tile_dim0 = 64
                16,                         // tile_dim1 = 16, tile_dim2 = 0
                64,                         // tensor_dim0_stride = 64 (lo32)
                0, 0 };
            tdm_int4 gz4 = { 0, 0, 0, 0 };
            tdm_int8 gz8 = { 0, 0, 0, 0, 0, 0, 0, 0 };
            // clang-23 / therock-10.0 6-arg form:
            // (g0, g1, g2, g3, extra_group, cpol)
            __builtin_amdgcn_tensor_load_to_lds(g0, g1, gz4, gz4, gz8, 0);
#if __has_builtin(__builtin_amdgcn_s_wait_tensorcnt)
            __builtin_amdgcn_s_wait_tensorcnt(0);
#else
            asm volatile("s_wait_tensorcnt 0x0" ::: "memory");
#endif
        }
#else
        {   // fallback: cooperative coalesced fill (512 dwords / 256 threads)
            const unsigned* src = (const unsigned*)(kb + (size_t)n0 * PROJ);
            unsigned* dst = (unsigned*)ldsK;
            int i0 = threadIdx.x, i1 = threadIdx.x + 256;
            dst[(i0 >> 5) * 36 + (i0 & 31)] = src[i0];
            dst[(i1 >> 5) * 36 + (i1 & 31)] = src[i1];
        }
#endif
        __syncthreads();
    }

    if (blockDead || m0 > n0 + 15) {            // wave-uniform; barrier already passed
        #pragma unroll
        for (int r = 0; r < 8; ++r) {
            int m = m0 + r + 8 * half;
            out[obase + (size_t)m * L_ + n] = NEGINF;
        }
        return;
    }

    const int rowA = m0 + nl;
    const int kbA  = 8 * half;
    const int kbB  = 8 * half;

    v8f c = {};
    #pragma unroll
    for (int kk = 0; kk < PROJ; kk += 32) {
        // A: q rows from global (bf16, already in register order)
        v8bf alo = *(const v8bf*)(qb + (size_t)rowA * PROJ + kk + kbA);
        v8bf ahi = *(const v8bf*)(qb + (size_t)rowA * PROJ + kk + 16 + kbA);
        v16bf a;
        #pragma unroll
        for (int i = 0; i < 8; ++i) { a[i] = alo[i]; a[8 + i] = ahi[i]; }

        // B: column n = K-row nl from LDS (padded stride 72, 16B aligned)
        const __bf16* slot = &ldsK[nl * 72 + kk + kbB];
        v8bf blo = *(const v8bf*)(slot);
        v8bf bhi = *(const v8bf*)(slot + 16);
        v16bf bm;
        #pragma unroll
        for (int i = 0; i < 8; ++i) { bm[i] = blo[i]; bm[8 + i] = bhi[i]; }

        c = __builtin_amdgcn_wmma_f32_16x16x32_bf16(
                false, a, false, bm, (short)0, c, false, false);
    }

    const bool keyok = amask[bb * L_ + n] != 0;
    #pragma unroll
    for (int r = 0; r < 8; ++r) {
        int m = m0 + r + 8 * half;
        float v = (keyok && (m <= n)) ? c[r] * 0.125f : NEGINF;
        out[obase + (size_t)m * L_ + n] = v;
    }
}

// =====================================================================
extern "C" void kernel_launch(void* const* d_in, const int* in_sizes, int n_in,
                              void* d_out, int out_size, void* d_ws, size_t ws_size,
                              hipStream_t stream) {
    const float* hs   = (const float*)d_in[0];   // (8,1024,1024) f32
    const int*   am   = (const int*)  d_in[1];   // (8,1024) i32
    const float* W    = (const float*)d_in[2];   // (1024,1152) f32
    const float* bias = (const float*)d_in[3];   // (1152,) f32
    float* out = (float*)d_out;                  // (8,9,1024,1024) f32

    unsigned short* qraw = (unsigned short*)d_ws;                 // 9.44 MB
    unsigned short* kraw = qraw + (size_t)B_ * NUM_LABELS * L_ * PROJ;

    dim3 g1(NTOT / 16, (B_ * L_) / 128, 1);            // (72, 64)
    proj_rope_kernel<<<g1, 256, 0, stream>>>(hs, W, bias, qraw, kraw);

    dim3 g2(L_ / 16, L_ / (16 * 8), B_ * NUM_LABELS);  // (64, 8, 72)
    qk_logits_kernel<<<g2, 256, 0, stream>>>(qraw, kraw, am, out);
}
